// SelfAttention_42365557407787
// MI455X (gfx1250) — compile-verified, matrix operand order checked by reading, not run on hardware
//
#include <hip/hip_runtime.h>
#include <hip/hip_bf16.h>

typedef _Float16 half_t;
typedef __attribute__((ext_vector_type(4)))  half_t v4h;
typedef __attribute__((ext_vector_type(8)))  half_t v8h;
typedef __attribute__((ext_vector_type(16))) half_t v16h;
typedef __attribute__((ext_vector_type(8)))  float  v8f;

#define B_N    4
#define S_LEN  2048
#define D_DIM  768
#define KCH    32              // K-dim per wmma
#define NKC_D  (D_DIM / KCH)   // 24
#define NKC_S  (S_LEN / KCH)   // 64

// Per-lane fragment load for 16-bit A/B operands of v_wmma_*_16x16x32.
// p must already point at: row_base + k_base + (lane>=16 ? 8 : 0).
// Lane holds two contiguous 8-half (16B) chunks at +0 and +16 halves.
__device__ __forceinline__ v16h frag_ld(const half_t* p) {
    v8h lo = *(const v8h*)(p);
    v8h hi = *(const v8h*)(p + 16);
    return __builtin_shufflevector(lo, hi, 0,1,2,3,4,5,6,7,8,9,10,11,12,13,14,15);
}

__device__ __forceinline__ v8f wmma_f16(v16h a, v16h b, v8f c) {
    return __builtin_amdgcn_wmma_f32_16x16x32_f16(false, a, false, b, (short)0, c, false, false);
}

// ---------------------------------------------------------------------------
// Kernel 1: f32 -> f16 conversion, 4 elems/thread (grid-stride)
// ---------------------------------------------------------------------------
__global__ void cvt_f16_kernel(const float* __restrict__ src,
                               half_t* __restrict__ dst, int n4) {
    int i = blockIdx.x * blockDim.x + threadIdx.x;
    int stride = gridDim.x * blockDim.x;
    for (; i < n4; i += stride) {
        float4 v = ((const float4*)src)[i];
        v4h h;
        h[0] = (half_t)v.x; h[1] = (half_t)v.y;
        h[2] = (half_t)v.z; h[3] = (half_t)v.w;
        *(v4h*)(dst + (size_t)i * 4) = h;
    }
}

// ---------------------------------------------------------------------------
// Kernel 2: QKV projection. grid = (D/64, B*S/16), block = 96 (3 waves).
// Each wave computes a 16x64 output block (4 accumulators, A-frag reused 4x).
// wave 0 -> Q (scaled by 1/sqrt(D), f16 row-major)
// wave 1 -> K (f16 row-major)
// wave 2 -> V (f16 transposed: Vt[b][d][s])
// ---------------------------------------------------------------------------
__global__ __launch_bounds__(96)
void qkv_kernel(const half_t* __restrict__ Xh,
                const half_t* __restrict__ Wqh,
                const half_t* __restrict__ Wkh,
                const half_t* __restrict__ Wvh,
                const float*  __restrict__ bq,
                const float*  __restrict__ bk,
                const float*  __restrict__ bv,
                half_t* __restrict__ Qh,
                half_t* __restrict__ Kh,
                half_t* __restrict__ Vt) {
    const int lane = threadIdx.x & 31;
    const int wave = threadIdx.x >> 5;          // 0=Q, 1=K, 2=V
    const int n0   = blockIdx.x * 64;           // output-channel block
    const int g0   = blockIdx.y * 16;           // global row in [0, B*S)
    const int r    = lane & 15;
    const int mhi  = (lane >> 4) ? 8 : 0;
    const int ko   = mhi;                       // k-offset pattern for frags

    const half_t* W  = (wave == 0) ? Wqh : (wave == 1) ? Wkh : Wvh;
    const float*  bb = (wave == 0) ? bq  : (wave == 1) ? bk  : bv;

    v8f acc[4];
    #pragma unroll
    for (int u = 0; u < 4; ++u) {
        float bias = bb[n0 + u * 16 + r];
        #pragma unroll
        for (int i = 0; i < 8; ++i) acc[u][i] = bias;
    }

    const half_t* xrow = Xh + (size_t)(g0 + r) * D_DIM + ko;
    const half_t* wrow = W  + (size_t)(n0 + r) * D_DIM + ko;  // +u*16 rows per chunk
    #pragma unroll
    for (int dk = 0; dk < NKC_D; ++dk) {
        v16h a = frag_ld(xrow + dk * KCH);
        #pragma unroll
        for (int u = 0; u < 4; ++u) {
            v16h wb = frag_ld(wrow + (size_t)u * 16 * D_DIM + dk * KCH);
            acc[u] = wmma_f16(a, wb, acc[u]);
        }
    }

    if (wave == 0) {
        const float qscale = 0.03608439182435161f;   // 1/sqrt(768)
        #pragma unroll
        for (int u = 0; u < 4; ++u) {
            half_t* dst = Qh + (size_t)g0 * D_DIM + n0 + u * 16 + r;
            #pragma unroll
            for (int i = 0; i < 8; ++i)
                dst[(size_t)(i + mhi) * D_DIM] = (half_t)(acc[u][i] * qscale);
        }
    } else if (wave == 1) {
        #pragma unroll
        for (int u = 0; u < 4; ++u) {
            half_t* dst = Kh + (size_t)g0 * D_DIM + n0 + u * 16 + r;
            #pragma unroll
            for (int i = 0; i < 8; ++i)
                dst[(size_t)(i + mhi) * D_DIM] = (half_t)acc[u][i];
        }
    } else {
        const int b  = g0 >> 11;                 // S = 2048
        const int s0 = g0 & (S_LEN - 1);
        #pragma unroll
        for (int u = 0; u < 4; ++u) {
            half_t* dst = Vt + ((size_t)b * D_DIM + n0 + u * 16 + r) * S_LEN + s0;
            #pragma unroll
            for (int i = 0; i < 8; ++i)
                dst[i + mhi] = (half_t)acc[u][i];
        }
    }
}

// ---------------------------------------------------------------------------
// Kernel 3: attention for one 16-row query tile.
// grid = B*S/16, block = 256 (8 waves -> 2 waves/SIMD for latency hiding).
// LDS: Qs 24KB | Sc 128KB (f32 scores) | Pq 64KB (f16 exp) | reductions ~1KB
// ---------------------------------------------------------------------------
#define SMEM_QS   (16 * D_DIM * 2)                 // 24576
#define SMEM_SC   (16 * S_LEN * 4)                 // 131072
#define SMEM_PQ   (16 * S_LEN * 2)                 // 65536
#define SMEM_RED  ((256 + 16 + 16) * 4)            // 1152
#define SMEM_TOT  (SMEM_QS + SMEM_SC + SMEM_PQ + SMEM_RED)

__global__ __launch_bounds__(256, 1)
void attn_kernel(const half_t* __restrict__ Qh,
                 const half_t* __restrict__ Kh,
                 const half_t* __restrict__ Vt,
                 float* __restrict__ Out) {
    extern __shared__ char smem[];
    half_t* Qs      = (half_t*)(smem);
    float*  Sc      = (float*)(smem + SMEM_QS);
    half_t* Pq      = (half_t*)(smem + SMEM_QS + SMEM_SC);
    float*  partial = (float*)(smem + SMEM_QS + SMEM_SC + SMEM_PQ); // [256]
    float*  rowmax  = partial + 256;                                // [16]
    float*  rowsum  = partial + 272;                                // [16]

    const int tid  = threadIdx.x;
    const int lane = tid & 31;
    const int wave = tid >> 5;                   // 0..7
    const int g0   = blockIdx.x * 16;            // global query row
    const int b    = g0 >> 11;
    const int r    = lane & 15;
    const int mhi  = (lane >> 4) ? 8 : 0;
    const int ko   = mhi;

    // ---- stage 0: stage Q tile (contiguous 24KB) into LDS ----
    {
        const uint4* src = (const uint4*)(Qh + (size_t)g0 * D_DIM);
        uint4* dst = (uint4*)Qs;
        #pragma unroll
        for (int i = 0; i < (SMEM_QS / 16) / 256; ++i)
            dst[tid + i * 256] = src[tid + i * 256];
    }
    __syncthreads();

    // ---- stage 1: scores = Q Kt (scale pre-folded into Q) ----
    // wave handles keys [wave*256, wave*256+256), 4 key-chunks per iteration:
    // one Q A-frag (LDS) feeds 4 K B-frags (global).
    for (int c = 0; c < 4; ++c) {
        const int key0 = wave * 256 + c * 64;
        v8f acc[4] = {};
        const half_t* qrow = Qs + r * D_DIM + ko;
        const half_t* krow = Kh + ((size_t)b * S_LEN + key0 + r) * D_DIM + ko;
        #pragma unroll
        for (int dk = 0; dk < NKC_D; ++dk) {
            v16h a = frag_ld(qrow + dk * KCH);
            #pragma unroll
            for (int u = 0; u < 4; ++u) {
                v16h kb = frag_ld(krow + (size_t)u * 16 * D_DIM + dk * KCH);
                acc[u] = wmma_f16(a, kb, acc[u]);
            }
        }
        #pragma unroll
        for (int u = 0; u < 4; ++u)
            #pragma unroll
            for (int i = 0; i < 8; ++i)
                Sc[(i + mhi) * S_LEN + key0 + u * 16 + r] = acc[u][i];
    }
    __syncthreads();

    // ---- stage 2: softmax stats + unnormalized exp(P) in f16 ----
    {
        const int row = tid >> 4;                // 16 rows, 16 threads/row
        const int seg = tid & 15;                // 128 cols/thread
        const float4* srow4 = (const float4*)(Sc + row * S_LEN + seg * 128);
        float m = -3.0e38f;
        for (int i = 0; i < 32; ++i) {
            float4 v = srow4[i];
            m = fmaxf(m, fmaxf(fmaxf(v.x, v.y), fmaxf(v.z, v.w)));
        }
        partial[tid] = m;
        __syncthreads();
        if (seg == 0) {
            float mm = partial[row * 16];
            #pragma unroll
            for (int i = 1; i < 16; ++i) mm = fmaxf(mm, partial[row * 16 + i]);
            rowmax[row] = mm;
        }
        __syncthreads();
        const float rm = rowmax[row];
        half_t* prow = Pq + row * S_LEN + seg * 128;
        float s = 0.0f;
        for (int i = 0; i < 32; ++i) {
            float4 v = srow4[i];
            float e0 = __expf(v.x - rm);
            float e1 = __expf(v.y - rm);
            float e2 = __expf(v.z - rm);
            float e3 = __expf(v.w - rm);
            s += (e0 + e1) + (e2 + e3);
            v4h h;
            h[0] = (half_t)e0; h[1] = (half_t)e1;
            h[2] = (half_t)e2; h[3] = (half_t)e3;
            *(v4h*)(prow + i * 4) = h;
        }
        partial[tid] = s;
        __syncthreads();
        if (seg == 0) {
            float ss = 0.0f;
            #pragma unroll
            for (int i = 0; i < 16; ++i) ss += partial[row * 16 + i];
            rowsum[row] = ss;
        }
        __syncthreads();
    }

    // ---- stage 3: Out = (P @ V) / rowsum ----
    // wave handles 96 d-cols as 2 iterations of 3 chunks: one P A-frag (LDS)
    // feeds 3 V B-frags (global).
    float rs[8];
    #pragma unroll
    for (int i = 0; i < 8; ++i) rs[i] = 1.0f / rowsum[i + mhi];

    for (int j = 0; j < 2; ++j) {
        const int n0 = wave * 96 + j * 48;
        v8f acc[3] = {};
        const half_t* prow = Pq + r * S_LEN + ko;
        const half_t* vrow = Vt + ((size_t)b * D_DIM + n0 + r) * S_LEN + ko;
        #pragma unroll 4
        for (int kc = 0; kc < NKC_S; ++kc) {
            v16h a = frag_ld(prow + kc * KCH);
            #pragma unroll
            for (int u = 0; u < 3; ++u) {
                v16h vb = frag_ld(vrow + (size_t)u * 16 * S_LEN + kc * KCH);
                acc[u] = wmma_f16(a, vb, acc[u]);
            }
        }
        #pragma unroll
        for (int u = 0; u < 3; ++u) {
            float* orow = Out + (size_t)g0 * D_DIM + n0 + u * 16 + r;
            #pragma unroll
            for (int i = 0; i < 8; ++i)
                orow[(size_t)(i + mhi) * D_DIM] = acc[u][i] * rs[i];
        }
    }
}

// ---------------------------------------------------------------------------
extern "C" void kernel_launch(void* const* d_in, const int* in_sizes, int n_in,
                              void* d_out, int out_size, void* d_ws, size_t ws_size,
                              hipStream_t stream) {
    const float* X  = (const float*)d_in[0];
    const float* Wq = (const float*)d_in[1];
    const float* bq = (const float*)d_in[2];
    const float* Wk = (const float*)d_in[3];
    const float* bk = (const float*)d_in[4];
    const float* Wv = (const float*)d_in[5];
    const float* bv = (const float*)d_in[6];
    float* out = (float*)d_out;

    const size_t nX = (size_t)B_N * S_LEN * D_DIM;     // 6291456
    const size_t nW = (size_t)D_DIM * D_DIM;           // 589824

    char* ws = (char*)d_ws;
    half_t* Xh  = (half_t*)ws;              ws += nX * sizeof(half_t);
    half_t* Wqh = (half_t*)ws;              ws += nW * sizeof(half_t);
    half_t* Wkh = (half_t*)ws;              ws += nW * sizeof(half_t);
    half_t* Wvh = (half_t*)ws;              ws += nW * sizeof(half_t);
    half_t* Qh  = (half_t*)ws;              ws += nX * sizeof(half_t);
    half_t* Kh  = (half_t*)ws;              ws += nX * sizeof(half_t);
    half_t* Vt  = (half_t*)ws;              ws += nX * sizeof(half_t);

    // 1) f32 -> f16 conversions (4 elems/thread)
    cvt_f16_kernel<<<1024, 256, 0, stream>>>(X,  Xh,  (int)(nX / 4));
    cvt_f16_kernel<<< 256, 256, 0, stream>>>(Wq, Wqh, (int)(nW / 4));
    cvt_f16_kernel<<< 256, 256, 0, stream>>>(Wk, Wkh, (int)(nW / 4));
    cvt_f16_kernel<<< 256, 256, 0, stream>>>(Wv, Wvh, (int)(nW / 4));

    // 2) QKV projection: 12 x 64-wide output blocks x 512 row tiles
    dim3 gq(D_DIM / 64, (B_N * S_LEN) / 16);
    qkv_kernel<<<gq, 96, 0, stream>>>(Xh, Wqh, Wkh, Wvh, bq, bk, bv, Qh, Kh, Vt);

    // 3) attention: one block per 16-row query tile, ~217KB dynamic LDS
    (void)hipFuncSetAttribute(reinterpret_cast<const void*>(attn_kernel),
                              hipFuncAttributeMaxDynamicSharedMemorySize,
                              SMEM_TOT);
    attn_kernel<<<(B_N * S_LEN) / 16, 256, SMEM_TOT, stream>>>(Qh, Kh, Vt, out);
}
